// AtomSelfInteracion_3985729650842
// MI455X (gfx1250) — matrix operand order repaired and checked by reading, not compiled, not run on hardware
//
#include <hip/hip_runtime.h>
#include <hip/hip_bf16.h>

// Problem constants (match reference)
#define N_ATOMS   50000
#define E_EDGES   200000
#define EMB_ATOM  256
#define EMB_EDGE  512
#define IN_SIZE   1024   // 2*EMB_ATOM + EMB_EDGE
#define OUT_SIZE  512

// Tiling
#define BM 128           // edges per workgroup
#define BN 128           // out-cols per workgroup
#define BK 32            // K chunk (one bf16 WMMA K)
#define A_STRIDE 40      // bf16 elems per LDS row (32 + 8 pad -> 80B stride)
#define B_STRIDE 40
#define KITERS   (IN_SIZE / BK)   // 32

typedef __bf16 bf16;
typedef __attribute__((ext_vector_type(16))) __bf16 v16bf;
typedef __attribute__((ext_vector_type(8)))  __bf16 bf16x8;
typedef __attribute__((ext_vector_type(8)))  float  v8f;

// Low 32 bits of a generic pointer to a __shared__ object are the wave-relative
// LDS byte address; that is what VDST of global_load_async_to_lds expects.
__device__ __forceinline__ unsigned lds_addr_of(const void* p) {
  return (unsigned)(unsigned long long)p;
}

__device__ __forceinline__ void async_copy_b128(const void* g, void* l) {
  asm volatile("global_load_async_to_lds_b128 %0, %1, off"
               :: "v"(lds_addr_of(l)), "v"(g)
               : "memory");
}

__device__ __forceinline__ void wait_async_all() {
  asm volatile("s_wait_asynccnt 0" ::: "memory");
}

// ---------------------------------------------------------------------------
// Kernel 0: W (1024x512 fp32, row-major k,n) -> Wt (512x1024 bf16, row-major n,k)
// ---------------------------------------------------------------------------
__global__ void wt_convert_kernel(const float* __restrict__ W, bf16* __restrict__ Wt) {
  int id = blockIdx.x * 256 + threadIdx.x;       // 0 .. 512*1024-1
  int n = id >> 10;
  int k = id & 1023;
  Wt[id] = (bf16)W[(size_t)k * OUT_SIZE + n];
}

// ---------------------------------------------------------------------------
// Kernel 1: fused gather + bf16 WMMA GEMM + SiLU
// ---------------------------------------------------------------------------
__global__ __launch_bounds__(256)
void AtomSelfInteracion_3985729650842_kernel(const float* __restrict__ h,
                                             const float* __restrict__ m_ij,
                                             const int*   __restrict__ idx_i,
                                             const int*   __restrict__ idx_j,
                                             const bf16*  __restrict__ Wt,
                                             float*       __restrict__ out)
{
  __shared__ bf16 Abuf[2][BM][A_STRIDE];   // 2*128*40*2 = 20480 B
  __shared__ bf16 Bbuf[2][BN][B_STRIDE];   // 2*128*40*2 = 20480 B

  const int t     = threadIdx.x;
  const int eBase = blockIdx.x * BM;
  const int nBase = blockIdx.y * BN;

  // ---- staging assignment: thread t owns (row = t>>1, 16-elem half = t&1) ----
  const int sRow  = t >> 1;      // 0..127 : edge-row for A, out-col for B
  const int sHalf = t & 1;

  int e  = eBase + sRow;
  int ec = e < E_EDGES ? e : (E_EDGES - 1);
  // NOTE: harness delivers integer inputs as 32-bit (values < N fit easily)
  const float* pHi = h    + (size_t)idx_i[ec] * EMB_ATOM + sHalf * 16;
  const float* pHj = h    + (size_t)idx_j[ec] * EMB_ATOM + sHalf * 16;
  const float* pM  = m_ij + (size_t)ec        * EMB_EDGE + sHalf * 16;

  // Walking pointers (reset at region boundaries with uniform compares,
  // avoiding a per-iteration select diamond on per-thread 64-bit pointers).
  const float* srcA = pHi;                                          // chunk 0
  const bf16*  srcB = Wt + (size_t)(nBase + sRow) * IN_SIZE + sHalf * 16;

  float4 fr[4];  // 16 fp32 of the A row-chunk in flight

  auto load_a_global = [&]() {
    fr[0] = *(const float4*)(srcA + 0);
    fr[1] = *(const float4*)(srcA + 4);
    fr[2] = *(const float4*)(srcA + 8);
    fr[3] = *(const float4*)(srcA + 12);
  };

  auto store_a_lds = [&](int buf) {
    bf16* d = &Abuf[buf][sRow][sHalf * 16];
    const float* f = (const float*)fr;
    bf16x8 o0, o1;
#pragma unroll
    for (int q = 0; q < 8; ++q) o0[q] = (bf16)f[q];
#pragma unroll
    for (int q = 0; q < 8; ++q) o1[q] = (bf16)f[8 + q];
    *(bf16x8*)(d)     = o0;
    *(bf16x8*)(d + 8) = o1;
  };

  auto stage_b = [&](int buf) {
    bf16* d = &Bbuf[buf][sRow][sHalf * 16];
    async_copy_b128(srcB,     d);        // 8 bf16 = 16 B each
    async_copy_b128(srcB + 8, d + 8);
  };

  // ---- compute assignment: 8 waves, wave = 32 rows x 64 cols (2x4 tiles) ----
  const int lane = t & 31;
  const int wid  = t >> 5;
  const int wm   = (wid & 3) * 32;
  const int wn   = (wid >> 2) * 64;
  const int lrow = lane & 15;
  const int hi   = lane >> 4;        // half-wave select
  const int kA   = hi ? 8 : 0;       // A lane K-base (ISA 16-bit A layout)
  const int kB   = hi ? 8 : 0;       // B lane K-base (same layout as A)

  v8f acc[2][4];
#pragma unroll
  for (int mi = 0; mi < 2; ++mi)
#pragma unroll
    for (int ni = 0; ni < 4; ++ni)
#pragma unroll
      for (int g = 0; g < 8; ++g) acc[mi][ni][g] = 0.0f;

  union VU { v16bf v; bf16x8 h2[2]; };

  auto compute = [&](int buf) {
    VU b[4];
#pragma unroll
    for (int ni = 0; ni < 4; ++ni) {
      const bf16* bp = &Bbuf[buf][wn + ni * 16 + lrow][kB];
      b[ni].h2[0] = *(const bf16x8*)(bp);
      b[ni].h2[1] = *(const bf16x8*)(bp + 16);
    }
#pragma unroll
    for (int mi = 0; mi < 2; ++mi) {
      const bf16* ap = &Abuf[buf][wm + mi * 16 + lrow][kA];
      VU a;
      a.h2[0] = *(const bf16x8*)(ap);        // K = kA .. kA+7
      a.h2[1] = *(const bf16x8*)(ap + 16);   // K = kA+16 .. kA+23
#pragma unroll
      for (int ni = 0; ni < 4; ++ni) {
        acc[mi][ni] = __builtin_amdgcn_wmma_f32_16x16x32_bf16(
            false, a.v, false, b[ni].v, (short)0, acc[mi][ni], false, false);
      }
    }
  };

  // ---- pipeline: prologue fills buffer 0 with chunk 0 ----
  load_a_global();
  stage_b(0);
  store_a_lds(0);
  srcA += BK;       // -> chunk 1 (still h_i region: 8 chunks per atom block)
  srcB += BK;

  // steady state: iterations 0..30 always prefetch chunk kc+1
  for (int kc = 0; kc < KITERS - 1; ++kc) {
    const int cur = kc & 1;
    wait_async_all();                // B chunk 'cur' landed in LDS
    __syncthreads();                 // everyone done with buffer cur^1

    load_a_global();                 // chunk kc+1 (overlaps WMMA below)
    stage_b(cur ^ 1);

    compute(cur);

    store_a_lds(cur ^ 1);

    // advance to chunk kc+2 (uniform, rarely-taken resets at region edges)
    srcA += BK;
    srcB += BK;
    if (kc + 2 == 8)  srcA = pHj;    // enter h[idx_j] region
    if (kc + 2 == 16) srcA = pM;     // enter m_ij region
  }

  // last chunk (KITERS-1 is odd -> buffer 1)
  wait_async_all();
  __syncthreads();
  compute((KITERS - 1) & 1);

  // ---- epilogue: SiLU + store (C layout: VGPR g -> row g / g+8) ----
  if (eBase + BM <= E_EDGES) {
    // full tile: no bound checks
#pragma unroll
    for (int mi = 0; mi < 2; ++mi) {
      const int rbase = eBase + wm + mi * 16 + hi * 8;
#pragma unroll
      for (int ni = 0; ni < 4; ++ni) {
        const int col = nBase + wn + ni * 16 + lrow;
#pragma unroll
        for (int g = 0; g < 8; ++g) {
          float c = acc[mi][ni][g];
          out[(size_t)(rbase + g) * OUT_SIZE + col] = c / (1.0f + __expf(-c));
        }
      }
    }
  } else {
#pragma unroll
    for (int mi = 0; mi < 2; ++mi) {
      const int rbase = eBase + wm + mi * 16 + hi * 8;
#pragma unroll
      for (int ni = 0; ni < 4; ++ni) {
        const int col = nBase + wn + ni * 16 + lrow;
#pragma unroll
        for (int g = 0; g < 8; ++g) {
          const int ee = rbase + g;
          if (ee < E_EDGES) {
            float c = acc[mi][ni][g];
            out[(size_t)ee * OUT_SIZE + col] = c / (1.0f + __expf(-c));
          }
        }
      }
    }
  }
}

// ---------------------------------------------------------------------------
extern "C" void kernel_launch(void* const* d_in, const int* in_sizes, int n_in,
                              void* d_out, int out_size, void* d_ws, size_t ws_size,
                              hipStream_t stream) {
  const float* h     = (const float*)d_in[0];
  const float* m_ij  = (const float*)d_in[1];
  const int*   idx_i = (const int*)d_in[2];
  const int*   idx_j = (const int*)d_in[3];
  const float* W     = (const float*)d_in[4];
  float*       out   = (float*)d_out;

  bf16* Wt = (bf16*)d_ws;  // 512*1024*2 = 1 MB of scratch

  wt_convert_kernel<<<(IN_SIZE * OUT_SIZE) / 256, 256, 0, stream>>>(W, Wt);

  dim3 grid((E_EDGES + BM - 1) / BM, OUT_SIZE / BN);   // (1563, 4)
  AtomSelfInteracion_3985729650842_kernel<<<grid, 256, 0, stream>>>(
      h, m_ij, idx_i, idx_j, Wt, out);
}
